// MultiTemporalAttention_4750233830138
// MI455X (gfx1250) — matrix operand order repaired and checked by reading, not compiled, not run on hardware
//
#include <hip/hip_runtime.h>
#include <hip/hip_fp16.h>

typedef __attribute__((ext_vector_type(16))) _Float16 v16h;
typedef __attribute__((ext_vector_type(8)))  _Float16 v8h;
typedef __attribute__((ext_vector_type(8)))  float    v8f;

#define Bq   4
#define Sq   2048
#define Dq   256
#define Hq   8
#define HSq  32
#define Rq   (Bq * Sq)   // 8192 token rows

#define AST  40   // LDS row stride (halves): 80B, multiple of 16B
#define BST  40
#define SST  36   // score patch row stride (floats): 144B, multiple of 16B
#define AROWS 130 // 128-row tile + 1-row halo each side (conv taps)

// branch-free tanh-GELU: 0.5*x*(1+tanh(z)) == x * sigmoid(2z)
__device__ inline float gelu_f(float x) {
    float z2 = 1.5957691216057308f * (x + 0.044715f * x * x * x);
    return x * (1.0f / (1.0f + __expf(-z2)));
}

__device__ inline v8f zero_v8f() {
    v8f z;
    for (int i = 0; i < 8; ++i) z[i] = 0.0f;
    return z;
}

__device__ inline v16h cat16(v8h lo, v8h hi) {
    return __builtin_shufflevector(lo, hi, 0, 1, 2, 3, 4, 5, 6, 7,
                                   8, 9, 10, 11, 12, 13, 14, 15);
}

__device__ inline v8h cvt8(const float* __restrict__ s) {
    v8h r;
    for (int u = 0; u < 8; ++u) r[u] = (_Float16)s[u];
    return r;
}

// ---------------------------------------------------------------------------
// LayerNorm over last dim (256): one block per row, wave-shuffle reduction
// ---------------------------------------------------------------------------
__global__ __launch_bounds__(256) void ln_kernel(const float* __restrict__ in,
                                                 const float* __restrict__ g,
                                                 const float* __restrict__ b,
                                                 float* __restrict__ out) {
    __shared__ float w1[8];
    __shared__ float w2[8];
    const int row  = blockIdx.x;
    const int t    = threadIdx.x;
    const int wave = t >> 5;
    const int lane = t & 31;
    float v  = in[(size_t)row * Dq + t];
    float a1 = v;
    float a2 = v * v;
    for (int m = 1; m < 32; m <<= 1) {
        a1 += __shfl_xor(a1, m, 32);
        a2 += __shfl_xor(a2, m, 32);
    }
    if (lane == 0) { w1[wave] = a1; w2[wave] = a2; }
    __syncthreads();
    float s1 = 0.0f, s2 = 0.0f;
    for (int i = 0; i < 8; ++i) { s1 += w1[i]; s2 += w2[i]; }
    float mean = s1 * (1.0f / Dq);
    float var  = s2 * (1.0f / Dq) - mean * mean;
    float inv  = rsqrtf(var + 1e-6f);
    out[(size_t)row * Dq + t] = (v - mean) * inv * g[t] + b[t];
}

// ---------------------------------------------------------------------------
// Conv1d(kernel-3 SAME)-as-GEMM / dense GEMM, ping-pong double buffered.
// TAPS/GELU/OUT16 are compile-time: tap loop fully unrolls, epilogue is
// straight-line.  Block: 8 waves, 128x64 tile, wave = 2x2 WMMA f16 tiles.
// ---------------------------------------------------------------------------
template <int TAPS, int GELU, int OUT16>
__global__ __launch_bounds__(256) void gemm_conv_kernel(
        const float* __restrict__ in, const float* __restrict__ Wt,
        const float* __restrict__ bias, const float* __restrict__ resid,
        float* __restrict__ out, _Float16* __restrict__ out16) {
    __shared__ alignas(16) _Float16 AhS[2 * AROWS * AST];      // [buf][row][k]
    __shared__ alignas(16) _Float16 BtS[2 * TAPS * 64 * BST];  // [buf][tap][n][k]

    const int tid  = threadIdx.x;
    const int wave = tid >> 5;
    const int lane = tid & 31;
    const int half = lane >> 4;
    const int l15  = lane & 15;
    const int wm   = wave & 3;    // 4 waves along M
    const int wn   = wave >> 2;   // 2 waves along N
    const int row0 = blockIdx.x * 128;
    const int n0   = blockIdx.y * 64;

    v8f acc[2][2];
    for (int mi = 0; mi < 2; ++mi)
        for (int nj = 0; nj < 2; ++nj)
            acc[mi][nj] = zero_v8f();

    // ---- staging helpers --------------------------------------------------
    auto stageA = [&](int buf, int kc) {
        if (tid < AROWS) {
            int sb = (row0 & (Sq - 1)) - 1 + tid;      // seq pos incl. halo
            _Float16* dst = &AhS[(buf * AROWS + tid) * AST];
            if (sb >= 0 && sb < Sq) {
                const float* src =
                    in + ((size_t)((row0 & ~(Sq - 1)) + sb)) * Dq + kc * 32;
                *(v8h*)(dst)      = cvt8(src);
                *(v8h*)(dst + 8)  = cvt8(src + 8);
                *(v8h*)(dst + 16) = cvt8(src + 16);
                *(v8h*)(dst + 24) = cvt8(src + 24);
            } else {
                v8h z;
                for (int u = 0; u < 8; ++u) z[u] = (_Float16)0.0f;
                *(v8h*)(dst)      = z;
                *(v8h*)(dst + 8)  = z;
                *(v8h*)(dst + 16) = z;
                *(v8h*)(dst + 24) = z;
            }
        }
    };
    auto stageB = [&](int buf, int kc) {
        int kb  = tid >> 3;            // 0..31
        int nn0 = (tid & 7) << 3;      // 0..56
#pragma unroll
        for (int t = 0; t < TAPS; ++t) {
            const float* src = Wt + (size_t)t * Dq * Dq +
                               (size_t)(kc * 32 + kb) * Dq + n0 + nn0;
            _Float16* bdst = &BtS[(buf * TAPS + t) * 64 * BST];
            for (int u = 0; u < 8; ++u)
                bdst[(nn0 + u) * BST + kb] = (_Float16)src[u];
        }
    };

    // ---- pipelined main loop over k-chunks --------------------------------
    stageA(0, 0);
    stageB(0, 0);
    for (int kc = 0; kc < Dq / 32; ++kc) {
        const int buf = kc & 1;
        __syncthreads();
        if (kc + 1 < Dq / 32) {        // prefetch next chunk into other buffer
            stageA(buf ^ 1, kc + 1);
            stageB(buf ^ 1, kc + 1);
        }
#pragma unroll
        for (int t = 0; t < TAPS; ++t) {
            const int ar = (TAPS == 3) ? t : 1;   // halo shift
            v16h af[2], bf[2];
            for (int mi = 0; mi < 2; ++mi) {
                const _Float16* ap =
                    &AhS[(buf * AROWS + wm * 32 + mi * 16 + l15 + ar) * AST];
                af[mi] = cat16(*(const v8h*)(ap + half * 8),
                               *(const v8h*)(ap + 16 + half * 8));
            }
            const _Float16* bbase = &BtS[(buf * TAPS + t) * 64 * BST];
            for (int nj = 0; nj < 2; ++nj) {
                const _Float16* bp =
                    bbase + (wn * 32 + nj * 16 + l15) * BST;
                bf[nj] = cat16(*(const v8h*)(bp + half * 8), *(const v8h*)(bp + 16 + half * 8));
            }
            for (int mi = 0; mi < 2; ++mi)
                for (int nj = 0; nj < 2; ++nj)
                    acc[mi][nj] = __builtin_amdgcn_wmma_f32_16x16x32_f16(
                        false, af[mi], false, bf[nj], (short)0, acc[mi][nj],
                        false, false);
        }
    }
    // ---- epilogue: bias (+gelu) (+residual), f32 or f16 output
    for (int mi = 0; mi < 2; ++mi) {
        for (int nj = 0; nj < 2; ++nj) {
            int col = n0 + wn * 32 + nj * 16 + l15;
            float bv = bias[col];
            for (int r = 0; r < 8; ++r) {
                int row = row0 + wm * 32 + mi * 16 + 8 * half + r;
                float val = acc[mi][nj][r] + bv;
                if (GELU) val = gelu_f(val);
                size_t idx = (size_t)row * Dq + col;
                if (resid) val += resid[idx];
                if (OUT16) out16[idx] = (_Float16)val;
                else       out[idx]   = val;
            }
        }
    }
}

// ---------------------------------------------------------------------------
// Flash attention over f16 q/k/v, ping-pong double buffered.
// grid = (S/64, H, B), block = 128 (4 waves x 16 q-rows).
// Softmax: wave writes its 16x32 score tile to an LDS patch; each lane then
// owns one row-half (contiguous 64B read), reduces locally, and only ONE
// shfl_xor(16) per max / per sum combines the halves.  Per-row stats live
// one-row-per-lane; rescale factors redistributed via a 16-float LDS scratch.
// ---------------------------------------------------------------------------
__global__ __launch_bounds__(128) void attn_kernel(
        const _Float16* __restrict__ q, const _Float16* __restrict__ k,
        const _Float16* __restrict__ v, const int* __restrict__ mask,
        const float* __restrict__ x, float* __restrict__ resid) {
    const int qb   = blockIdx.x;
    const int head = blockIdx.y;
    const int b    = blockIdx.z;
    const int tid  = threadIdx.x;
    const int wave = tid >> 5;
    const int lane = tid & 31;
    const int half = lane >> 4;
    const int l15  = lane & 15;

    __shared__ alignas(16) _Float16 KhS[2 * 32 * AST];     // [buf][key][hs]
    __shared__ alignas(16) _Float16 VtS[2 * 32 * AST];     // [buf][hs][key]
    __shared__ alignas(16) _Float16 Pl[4 * 16 * AST];      // per-wave [m][key]
    __shared__ alignas(16) float    ScS[4 * 16 * SST];     // per-wave [m][col]
    __shared__ alignas(16) float    sfb[4 * 16];           // per-wave row scratch
    __shared__ float maskadd[2][32];

    auto stageKV = [&](int buf, int jb) {
        int key = tid >> 2;            // 0..31
        int h0  = (tid & 3) * 8;       // 0,8,16,24
        size_t base = ((size_t)(b * Sq + jb * 32 + key)) * Dq + head * HSq;
        *(v8h*)&KhS[(buf * 32 + key) * AST + h0] = *(const v8h*)(k + base + h0);
        v8h vv = *(const v8h*)(v + base + h0);
        for (int u = 0; u < 8; ++u)
            VtS[(buf * 32 + h0 + u) * AST + key] = vv[u];
        if (tid < 32)
            maskadd[buf][tid] = mask[b * Sq + jb * 32 + tid] ? 0.0f : -1e30f;
    };

    // --- Q A-fragment with 1/sqrt(HS) folded in, resident all sweep
    const float scale = 0.17677669529663687f;   // 1/sqrt(32)
    const int srow = qb * 64 + wave * 16 + l15;
    const _Float16* qrow = q + ((size_t)(b * Sq + srow)) * Dq + head * HSq;
    v8h q0 = *(const v8h*)(qrow + half * 8);
    v8h q1 = *(const v8h*)(qrow + 16 + half * 8);
    for (int u = 0; u < 8; ++u) {
        q0[u] = (_Float16)((float)q0[u] * scale);
        q1[u] = (_Float16)((float)q1[u] * scale);
    }
    v16h qf = cat16(q0, q1);

    v8f oacc0 = zero_v8f();
    v8f oacc1 = zero_v8f();
    float mrow = -1e30f;   // stats for row l15 (replicated across halves)
    float lrow = 0.0f;

    float* scw = &ScS[wave * 16 * SST];
    _Float16* pw = &Pl[wave * 16 * AST];
    float* sfw = &sfb[wave * 16];

    stageKV(0, 0);
    for (int jb = 0; jb < Sq / 32; ++jb) {
        const int buf = jb & 1;
        __syncthreads();
        if (jb + 1 < Sq / 32) stageKV(buf ^ 1, jb + 1);

        // ---- scores = Qs * K^T  (B-frag = K rows, contiguous)
        v16h bk0, bk1;
        {
            const _Float16* kp = &KhS[(buf * 32) * AST];
            const _Float16* p0 = kp + l15 * AST + half * 8;
            const _Float16* p1 = kp + (16 + l15) * AST + half * 8;
            bk0 = cat16(*(const v8h*)p0, *(const v8h*)(p0 + 16));
            bk1 = cat16(*(const v8h*)p1, *(const v8h*)(p1 + 16));
        }
        v8f s0 = zero_v8f(), s1 = zero_v8f();
        s0 = __builtin_amdgcn_wmma_f32_16x16x32_f16(false, qf, false, bk0,
                                                    (short)0, s0, false, false);
        s1 = __builtin_amdgcn_wmma_f32_16x16x32_f16(false, qf, false, bk1,
                                                    (short)0, s1, false, false);
        float ma0 = maskadd[buf][l15];
        float ma1 = maskadd[buf][16 + l15];

        // ---- dump masked scores to per-wave patch (C-layout)
        for (int r = 0; r < 8; ++r) {
            scw[(8 * half + r) * SST + l15]      = s0[r] + ma0;
            scw[(8 * half + r) * SST + 16 + l15] = s1[r] + ma1;
        }

        // ---- row-per-lane softmax: lane owns row l15, cols half*16..+15
        const float* srp = &scw[l15 * SST + half * 16];
        float sv[16];
        for (int u = 0; u < 16; ++u) sv[u] = srp[u];
        float vmax = sv[0];
        for (int u = 1; u < 16; ++u) vmax = fmaxf(vmax, sv[u]);
        vmax = fmaxf(vmax, __shfl_xor(vmax, 16, 32));
        float mnew = fmaxf(mrow, vmax);
        float sf = __expf(mrow - mnew);
        float rs = 0.0f;
        v8h pv0, pv1;
        for (int u = 0; u < 8; ++u) {
            float e = __expf(sv[u] - mnew);
            rs += e;
            pv0[u] = (_Float16)e;
        }
        for (int u = 0; u < 8; ++u) {
            float e = __expf(sv[8 + u] - mnew);
            rs += e;
            pv1[u] = (_Float16)e;
        }
        rs += __shfl_xor(rs, 16, 32);
        lrow = lrow * sf + rs;
        mrow = mnew;

        // ---- P already row-major per lane: 2 aligned v8h stores
        *(v8h*)&pw[l15 * AST + half * 16]     = pv0;
        *(v8h*)&pw[l15 * AST + half * 16 + 8] = pv1;

        // ---- redistribute sf to C-layout and rescale O
        if (half == 0) sfw[l15] = sf;
        for (int r = 0; r < 8; ++r) {
            float sc = sfw[8 * half + r];
            oacc0[r] *= sc;
            oacc1[r] *= sc;
        }

        // ---- A-frag of P, then O += P * V  (B-frag = Vt rows, contiguous)
        v16h pa;
        {
            const _Float16* pp = &pw[l15 * AST];
            pa = cat16(*(const v8h*)(pp + half * 8),
                       *(const v8h*)(pp + 16 + half * 8));
        }
        v16h bv0, bv1;
        {
            const _Float16* vp = &VtS[(buf * 32) * AST];
            const _Float16* p0v = vp + l15 * AST + half * 8;
            const _Float16* p1v = vp + (16 + l15) * AST + half * 8;
            bv0 = cat16(*(const v8h*)p0v, *(const v8h*)(p0v + 16));
            bv1 = cat16(*(const v8h*)p1v, *(const v8h*)(p1v + 16));
        }
        oacc0 = __builtin_amdgcn_wmma_f32_16x16x32_f16(false, pa, false, bv0,
                                                       (short)0, oacc0, false, false);
        oacc1 = __builtin_amdgcn_wmma_f32_16x16x32_f16(false, pa, false, bv1,
                                                       (short)0, oacc1, false, false);
    }

    // ---- redistribute 1/l to C-layout, normalize, fuse residual
    if (half == 0) sfw[l15] = lrow;
    for (int r = 0; r < 8; ++r) {
        int s = qb * 64 + wave * 16 + 8 * half + r;
        float invl = 1.0f / sfw[8 * half + r];
        size_t base = ((size_t)(b * Sq + s)) * Dq + head * HSq;
        resid[base + l15]      = x[base + l15]      + oacc0[r] * invl;
        resid[base + 16 + l15] = x[base + 16 + l15] + oacc1[r] * invl;
    }
}

// ---------------------------------------------------------------------------
extern "C" void kernel_launch(void* const* d_in, const int* in_sizes, int n_in,
                              void* d_out, int out_size, void* d_ws, size_t ws_size,
                              hipStream_t stream) {
    (void)in_sizes; (void)n_in; (void)out_size; (void)ws_size;
    const float* x     = (const float*)d_in[0];
    const float* ln1_g = (const float*)d_in[1];
    const float* ln1_b = (const float*)d_in[2];
    const float* Wq    = (const float*)d_in[3];
    const float* bq    = (const float*)d_in[4];
    const float* Wk    = (const float*)d_in[5];
    const float* bk    = (const float*)d_in[6];
    const float* Wv    = (const float*)d_in[7];
    const float* bv    = (const float*)d_in[8];
    const float* ln2_g = (const float*)d_in[9];
    const float* ln2_b = (const float*)d_in[10];
    const float* Wo1   = (const float*)d_in[11];
    const float* bo1   = (const float*)d_in[12];
    const float* Wo2   = (const float*)d_in[13];
    const float* bo2   = (const float*)d_in[14];
    const int*   mask  = (const int*)d_in[15];
    float* out = (float*)d_out;

    const size_t NB = (size_t)Rq * Dq;          // 2M elements per buffer
    float*     h     = (float*)d_ws;            // 8 MB
    float*     tmp   = h + NB;                  // 8 MB
    float*     resid = tmp + NB;                // 8 MB
    _Float16*  q16   = (_Float16*)(resid + NB); // 4 MB
    _Float16*  k16   = q16 + NB;                // 4 MB
    _Float16*  v16   = k16 + NB;                // 4 MB

    const size_t TAPW = (size_t)3 * Dq * Dq;    // per-layer conv weight stride
    dim3 gG(Rq / 128, Dq / 64), bG(256);

    // h = LN1(x)
    ln_kernel<<<Rq, 256, 0, stream>>>(x, ln1_g, ln1_b, h);

    // q / k / v temporal encoders: conv3 -> gelu -> conv3 (f16 out)
    gemm_conv_kernel<3, 1, 0><<<gG, bG, 0, stream>>>(h,   Wq,        bq,      nullptr, tmp, nullptr);
    gemm_conv_kernel<3, 0, 1><<<gG, bG, 0, stream>>>(tmp, Wq + TAPW, bq + Dq, nullptr, nullptr, q16);
    gemm_conv_kernel<3, 1, 0><<<gG, bG, 0, stream>>>(h,   Wk,        bk,      nullptr, tmp, nullptr);
    gemm_conv_kernel<3, 0, 1><<<gG, bG, 0, stream>>>(tmp, Wk + TAPW, bk + Dq, nullptr, nullptr, k16);
    gemm_conv_kernel<3, 1, 0><<<gG, bG, 0, stream>>>(h,   Wv,        bv,      nullptr, tmp, nullptr);
    gemm_conv_kernel<3, 0, 1><<<gG, bG, 0, stream>>>(tmp, Wv + TAPW, bv + Dq, nullptr, nullptr, v16);

    // attention with fused residual: resid = x + attn
    attn_kernel<<<dim3(Sq / 64, Hq, Bq), 128, 0, stream>>>(q16, k16, v16,
                                                           mask, x, resid);

    // output MLP: out = gelu(LN2(resid) @ Wo1 + bo1) @ Wo2 + bo2 + resid
    ln_kernel<<<Rq, 256, 0, stream>>>(resid, ln2_g, ln2_b, h);
    gemm_conv_kernel<1, 1, 0><<<gG, bG, 0, stream>>>(h,   Wo1, bo1, nullptr, tmp, nullptr);
    gemm_conv_kernel<1, 0, 0><<<gG, bG, 0, stream>>>(tmp, Wo2, bo2, resid,   out, nullptr);
}
